// TransformerBlock_88983132439184
// MI455X (gfx1250) — matrix operand order repaired and checked by reading, not compiled, not run on hardware
//
#include <hip/hip_runtime.h>
#include <stdint.h>

// ---------------------------------------------------------------------------
// Types
// ---------------------------------------------------------------------------
typedef __bf16 bf16_t;
typedef __bf16 bf16x16 __attribute__((ext_vector_type(16)));
typedef __bf16 bf16x8  __attribute__((ext_vector_type(8)));
typedef float  f32x8   __attribute__((ext_vector_type(8)));

static __device__ __forceinline__ bf16_t f2bf_ok(float f) {
    unsigned u = __builtin_bit_cast(unsigned, f);
    unsigned r = (u + 0x7FFFu + ((u >> 16) & 1u)) >> 16;   // round-nearest-even
    unsigned short s = (unsigned short)r;
    return __builtin_bit_cast(bf16_t, s);
}

// ---------------------------------------------------------------------------
// Fragment loaders (CDNA5 WMMA 16x16x32 bf16 register layouts, ISA 7.12.2)
//   A (16x32, MxK): lane 0-15 -> M=lane, K={0..7,16..23}; lane 16-31 -> K={8..15,24..31}
//   B (32x16, KxN): lane 0-15 -> N=lane, K={0..7,16..23}; lane 16-31 -> K={8..15,24..31}
// ---------------------------------------------------------------------------
static __device__ __forceinline__ bf16x16 load_frag_a(const bf16_t* base, int lane, int stride) {
    int row = lane & 15;
    int kc  = (lane & 16) ? 8 : 0;
    const bf16_t* p = base + (size_t)row * stride;
    bf16x8 lo = *(const bf16x8*)(p + kc);
    bf16x8 hi = *(const bf16x8*)(p + kc + 16);
    bf16x16 a;
#pragma unroll
    for (int i = 0; i < 8; ++i) { a[i] = lo[i]; a[i + 8] = hi[i]; }
    return a;
}

static __device__ __forceinline__ bf16x16 load_frag_b(const bf16_t* base, int lane, long long strideN) {
    int col = lane & 15;
    int kc  = (lane & 16) ? 8 : 0;
    const bf16_t* p = base + (size_t)col * strideN + kc;
    bf16x8 lo = *(const bf16x8*)(p);
    bf16x8 hi = *(const bf16x8*)(p + 16);
    bf16x16 b;
#pragma unroll
    for (int i = 0; i < 8; ++i) { b[i] = lo[i]; b[i + 8] = hi[i]; }
    return b;
}

// Async DMA: one 16B chunk per lane, global -> LDS, tracked by ASYNCcnt.
// LDS byte address = low 32 bits of the generic pointer (aperture truncation).
static __device__ __forceinline__ void async_copy_b128(void* lds_dst, const void* gsrc) {
    unsigned       ldsa = (unsigned)(uintptr_t)lds_dst;
    unsigned long long ga = (unsigned long long)(uintptr_t)gsrc;
    asm volatile("global_load_async_to_lds_b128 %0, %1, off"
                 :: "v"(ldsa), "v"(ga) : "memory");
}

// ---------------------------------------------------------------------------
// Elementwise convert f32 -> bf16
// ---------------------------------------------------------------------------
__global__ void cvt_bf16_kernel(const float* __restrict__ in, bf16_t* __restrict__ out, long long n) {
    long long i = ((long long)blockIdx.x * blockDim.x + threadIdx.x) * 4;
    if (i + 3 < n) {
#pragma unroll
        for (int k = 0; k < 4; ++k) out[i + k] = f2bf_ok(in[i + k]);
    }
}

// W[K,N] f32 (row-major) -> Wt[N,K] bf16 (row-major)
__global__ void transpose_cvt_kernel(const float* __restrict__ in, bf16_t* __restrict__ out,
                                     int K, int N) {
    long long idx = (long long)blockIdx.x * blockDim.x + threadIdx.x;
    long long total = (long long)K * N;
    if (idx < total) {
        int k = (int)(idx / N);
        int n = (int)(idx % N);
        out[(size_t)n * K + k] = f2bf_ok(in[idx]);
    }
}

// ---------------------------------------------------------------------------
// Tiled WMMA GEMM:  C[M,N] = A[M,K](bf16,row) * Bt[N,K](bf16,row)^T + bias
// Workgroup tile 128x64, BK=32, 8 waves (4x2), each wave 32x32 (2x2 frags).
// Tiles staged with GLOBAL_LOAD_ASYNC_TO_LDS_B128 (ASYNCcnt), no VGPR bounce.
// mode: 0 = bf16 out, 1 = f32 out, 2 = bf16 out transposed into Vt[b,h,dk,t],
//       3 = bf16 out + ReLU
// ---------------------------------------------------------------------------
#define BM 128
#define BN 64
#define BK 32
#define LDSA 40   // padded row stride (elements), keeps 16B alignment

__global__ __launch_bounds__(256)
void gemm_bf16_kernel(const bf16_t* __restrict__ A, const bf16_t* __restrict__ Bt,
                      const float* __restrict__ bias, void* __restrict__ out,
                      int M, int N, int K, int mode, int T, int H) {
    __shared__ bf16_t As[BM * LDSA];
    __shared__ bf16_t Bs[BN * LDSA];

    const int tid  = threadIdx.x;
    const int lane = tid & 31;
    const int wid  = tid >> 5;
    const int wm   = wid & 3;      // 0..3
    const int wn   = wid >> 2;     // 0..1

    const int tilesM = M / BM;
    const int tm = blockIdx.x % tilesM;
    const int tn = blockIdx.x / tilesM;
    const int m0 = tm * BM;
    const int n0 = tn * BN;

    // per-thread staging decode (256 thr): A = 512 chunks (2/thr), B = 256 chunks
    const int rowC = tid >> 2;            // 0..63
    const int k8C  = (tid & 3) * 8;       // 0,8,16,24
    bf16_t* ldsA0 = &As[rowC * LDSA + k8C];
    bf16_t* ldsA1 = &As[(rowC + 64) * LDSA + k8C];
    bf16_t* ldsB  = &Bs[rowC * LDSA + k8C];

    f32x8 acc[2][2];
#pragma unroll
    for (int mi = 0; mi < 2; ++mi)
#pragma unroll
        for (int ni = 0; ni < 2; ++ni)
#pragma unroll
            for (int r = 0; r < 8; ++r) acc[mi][ni][r] = 0.0f;

    for (int k0 = 0; k0 < K; k0 += BK) {
        // stage A(128x32) + B(64x32) tiles: async DMA straight into LDS
        async_copy_b128(ldsA0, &A[(size_t)(m0 + rowC) * K + k0 + k8C]);
        async_copy_b128(ldsA1, &A[(size_t)(m0 + rowC + 64) * K + k0 + k8C]);
        async_copy_b128(ldsB,  &Bt[(size_t)(n0 + rowC) * K + k0 + k8C]);
        if (k0 + BK < K) {  // hint next A tile into L2 (global_prefetch_b8)
            __builtin_prefetch(&A[(size_t)(m0 + (tid >> 1)) * K + k0 + BK], 0, 3);
        }
        asm volatile("s_wait_asynccnt 0x0" ::: "memory");
        __syncthreads();

        bf16x16 af[2], bff[2];
#pragma unroll
        for (int mi = 0; mi < 2; ++mi)
            af[mi] = load_frag_a(&As[(wm * 32 + mi * 16) * LDSA], lane, LDSA);
#pragma unroll
        for (int ni = 0; ni < 2; ++ni)
            bff[ni] = load_frag_b(&Bs[(wn * 32 + ni * 16) * LDSA], lane, LDSA);

#pragma unroll
        for (int mi = 0; mi < 2; ++mi)
#pragma unroll
            for (int ni = 0; ni < 2; ++ni)
                acc[mi][ni] = __builtin_amdgcn_wmma_f32_16x16x32_bf16(
                    false, af[mi], false, bff[ni], (short)0, acc[mi][ni], false, false);
        __syncthreads();
    }

    // epilogue
#pragma unroll
    for (int mi = 0; mi < 2; ++mi) {
#pragma unroll
        for (int ni = 0; ni < 2; ++ni) {
#pragma unroll
            for (int r = 0; r < 8; ++r) {
                int m = m0 + wm * 32 + mi * 16 + r + ((lane & 16) ? 8 : 0);
                int n = n0 + wn * 32 + ni * 16 + (lane & 15);
                float v = acc[mi][ni][r] + bias[n];
                if (mode == 3) v = fmaxf(v, 0.0f);
                if (mode == 1) {
                    ((float*)out)[(size_t)m * N + n] = v;
                } else if (mode == 2) {
                    int b = m / T, t = m % T;
                    int hh = n >> 6, dkc = n & 63;
                    ((bf16_t*)out)[(((size_t)b * H + hh) * 64 + dkc) * T + t] = f2bf_ok(v);
                } else {
                    ((bf16_t*)out)[(size_t)m * N + n] = f2bf_ok(v);
                }
            }
        }
    }
}

// ---------------------------------------------------------------------------
// Flash attention (no mask). Each wave: 16 query rows of one (b,h); loops over
// 32-wide KV blocks. QK^T and P*V via v_wmma_f32_16x16x32_bf16. P converted
// C-layout -> A-layout through per-wave LDS (bf16) with s_wait_dscnt.
//   Q,K : bf16 [B*T, D] (head h at column h*64), Vt : bf16 [B*H, 64, T]
//   ctx : bf16 [B*T, D]
// ---------------------------------------------------------------------------
__global__ __launch_bounds__(256)
void attn_kernel(const bf16_t* __restrict__ Q, const bf16_t* __restrict__ Kb,
                 const bf16_t* __restrict__ Vt, bf16_t* __restrict__ ctx,
                 int T, int H, int D) {
    __shared__ bf16_t Pl[8 * 16 * LDSA];

    const int lane = threadIdx.x & 31;
    const int wid  = threadIdx.x >> 5;
    const int tile = blockIdx.x * 8 + wid;
    const int rtPer = T / 16;
    const int bh = tile / rtPer;
    const int rt = tile % rtPer;
    const int b = bh / H, h = bh % H;
    const int q0 = rt * 16;
    bf16_t* Pw = &Pl[wid * 16 * LDSA];

    const bf16_t* Qbase = Q + (size_t)(b * T + q0) * D + h * 64;
    bf16x16 aQ0 = load_frag_a(Qbase,      lane, D);   // dk 0..31
    bf16x16 aQ1 = load_frag_a(Qbase + 32, lane, D);   // dk 32..63

    f32x8 acc[4];
#pragma unroll
    for (int n = 0; n < 4; ++n)
#pragma unroll
        for (int r = 0; r < 8; ++r) acc[n][r] = 0.0f;

    float mrow[8], lrow[8];
#pragma unroll
    for (int r = 0; r < 8; ++r) { mrow[r] = -3.0e38f; lrow[r] = 0.0f; }

    const float sm_scale = 0.125f;   // 1/sqrt(64)

    for (int kv0 = 0; kv0 < T; kv0 += 32) {
        // ---- S = Q K^T (16 x 32) ----
        f32x8 s0, s1;
#pragma unroll
        for (int r = 0; r < 8; ++r) { s0[r] = 0.0f; s1[r] = 0.0f; }
        const bf16_t* Kbase = Kb + (size_t)(b * T + kv0) * D + h * 64;
#pragma unroll
        for (int kt = 0; kt < 2; ++kt) {
            bf16x16 aq = kt ? aQ1 : aQ0;
            bf16x16 bk0 = load_frag_b(Kbase + kt * 32,                  lane, D);
            bf16x16 bk1 = load_frag_b(Kbase + 16 * (size_t)D + kt * 32, lane, D);
            s0 = __builtin_amdgcn_wmma_f32_16x16x32_bf16(false, aq, false, bk0, (short)0, s0, false, false);
            s1 = __builtin_amdgcn_wmma_f32_16x16x32_bf16(false, aq, false, bk1, (short)0, s1, false, false);
        }

        // ---- online softmax over the 32 columns ----
#pragma unroll
        for (int r = 0; r < 8; ++r) {
            float v0 = s0[r] * sm_scale;
            float v1 = s1[r] * sm_scale;
            float mx = fmaxf(v0, v1);
#pragma unroll
            for (int msk = 1; msk < 16; msk <<= 1)
                mx = fmaxf(mx, __shfl_xor(mx, msk, 32));
            float mnew = fmaxf(mrow[r], mx);
            float sc = __expf(mrow[r] - mnew);
            mrow[r] = mnew;
            float p0 = __expf(v0 - mnew);
            float p1 = __expf(v1 - mnew);
            float rs = p0 + p1;
#pragma unroll
            for (int msk = 1; msk < 16; msk <<= 1)
                rs += __shfl_xor(rs, msk, 32);
            lrow[r] = lrow[r] * sc + rs;
#pragma unroll
            for (int n = 0; n < 4; ++n) acc[n][r] *= sc;
            int prow = r + ((lane & 16) ? 8 : 0);
            int pcol = lane & 15;
            Pw[prow * LDSA + pcol]      = f2bf_ok(p0);
            Pw[prow * LDSA + pcol + 16] = f2bf_ok(p1);
        }
        asm volatile("s_wait_dscnt 0x0" ::: "memory");

        // ---- ctx += P * V ----
        bf16x16 aP = load_frag_a(Pw, lane, LDSA);
#pragma unroll
        for (int n = 0; n < 4; ++n) {
            const bf16_t* Vb = Vt + ((size_t)bh * 64 + n * 16) * T + kv0;
            bf16x16 bv = load_frag_b(Vb, lane, T);
            acc[n] = __builtin_amdgcn_wmma_f32_16x16x32_bf16(false, aP, false, bv, (short)0, acc[n], false, false);
        }
    }

    // ---- normalize and store ctx (bf16, [B*T, D] layout) ----
#pragma unroll
    for (int r = 0; r < 8; ++r) {
        float inv = 1.0f / lrow[r];
        int row = q0 + r + ((lane & 16) ? 8 : 0);
        size_t base = (size_t)(b * T + row) * D + h * 64 + (lane & 15);
#pragma unroll
        for (int n = 0; n < 4; ++n)
            ctx[base + n * 16] = f2bf_ok(acc[n][r] * inv);
    }
}

// ---------------------------------------------------------------------------
// Fused residual + LayerNorm. One block (256 thr) per row of D=1024.
// t = resid + y ; out = (t - mu)/sqrt(var+eps) * g + be  (f32, optional bf16)
// ---------------------------------------------------------------------------
__global__ __launch_bounds__(256)
void ln_kernel(const float* __restrict__ resid, const float* __restrict__ y,
               const float* __restrict__ g, const float* __restrict__ be,
               float* __restrict__ outf, bf16_t* __restrict__ outb, int D) {
    __shared__ float s1[256], s2[256];
    const int row = blockIdx.x;
    const int tid = threadIdx.x;
    float t[4];
    float sum = 0.0f, sq = 0.0f;
#pragma unroll
    for (int k = 0; k < 4; ++k) {
        int i = tid * 4 + k;
        t[k] = resid[(size_t)row * D + i] + y[(size_t)row * D + i];
        sum += t[k];
        sq  += t[k] * t[k];
    }
    s1[tid] = sum; s2[tid] = sq;
    __syncthreads();
    for (int s = 128; s > 0; s >>= 1) {
        if (tid < s) { s1[tid] += s1[tid + s]; s2[tid] += s2[tid + s]; }
        __syncthreads();
    }
    float mu  = s1[0] / D;
    float var = s2[0] / D - mu * mu;
    float rstd = rsqrtf(var + 1e-5f);
#pragma unroll
    for (int k = 0; k < 4; ++k) {
        int i = tid * 4 + k;
        float v = (t[k] - mu) * rstd * g[i] + be[i];
        outf[(size_t)row * D + i] = v;
        if (outb) outb[(size_t)row * D + i] = f2bf_ok(v);
    }
}

// ---------------------------------------------------------------------------
// Host orchestration
// ---------------------------------------------------------------------------
extern "C" void kernel_launch(void* const* d_in, const int* in_sizes, int n_in,
                              void* d_out, int out_size, void* d_ws, size_t ws_size,
                              hipStream_t stream) {
    (void)in_sizes; (void)n_in; (void)out_size; (void)ws_size;
    const int B = 4, T = 2048, D = 1024, H = 16, F = 4096;
    const int M = B * T;

    const float* x  = (const float*)d_in[0];
    const float* Wq = (const float*)d_in[1];  const float* bq = (const float*)d_in[2];
    const float* Wk = (const float*)d_in[3];  const float* bk = (const float*)d_in[4];
    const float* Wv = (const float*)d_in[5];  const float* bv = (const float*)d_in[6];
    const float* Wo = (const float*)d_in[7];  const float* bo = (const float*)d_in[8];
    const float* g1 = (const float*)d_in[9];  const float* be1 = (const float*)d_in[10];
    const float* W1 = (const float*)d_in[11]; const float* b1 = (const float*)d_in[12];
    const float* W2 = (const float*)d_in[13]; const float* b2 = (const float*)d_in[14];
    const float* g2 = (const float*)d_in[15]; const float* be2 = (const float*)d_in[16];

    // workspace bump allocator
    char* ws = (char*)d_ws;
    size_t off = 0;
    auto alloc = [&](size_t bytes) -> char* {
        char* p = ws + off;
        off = (off + bytes + 255) & ~(size_t)255;
        return p;
    };
    bf16_t* xb   = (bf16_t*)alloc((size_t)M * D * 2);
    bf16_t* Wqt  = (bf16_t*)alloc((size_t)D * D * 2);
    bf16_t* Wkt  = (bf16_t*)alloc((size_t)D * D * 2);
    bf16_t* Wvt  = (bf16_t*)alloc((size_t)D * D * 2);
    bf16_t* Wot  = (bf16_t*)alloc((size_t)D * D * 2);
    bf16_t* W1t  = (bf16_t*)alloc((size_t)D * F * 2);
    bf16_t* W2t  = (bf16_t*)alloc((size_t)F * D * 2);
    bf16_t* Qb   = (bf16_t*)alloc((size_t)M * D * 2);
    bf16_t* Kbuf = (bf16_t*)alloc((size_t)M * D * 2);
    bf16_t* Vt   = (bf16_t*)alloc((size_t)M * D * 2);
    bf16_t* ctxb = (bf16_t*)alloc((size_t)M * D * 2);
    float*  raw  = (float*) alloc((size_t)M * D * 4);   // reused: attn_out, then ffn_out
    float*  x1f  = (float*) alloc((size_t)M * D * 4);
    bf16_t* x1b  = (bf16_t*)alloc((size_t)M * D * 2);
    bf16_t* hb   = (bf16_t*)alloc((size_t)M * F * 2);

    // 1) precision / layout prep
    cvt_bf16_kernel<<<((long long)M * D) / (256 * 4), 256, 0, stream>>>(x, xb, (long long)M * D);
    transpose_cvt_kernel<<<(D * D) / 256, 256, 0, stream>>>(Wq, Wqt, D, D);
    transpose_cvt_kernel<<<(D * D) / 256, 256, 0, stream>>>(Wk, Wkt, D, D);
    transpose_cvt_kernel<<<(D * D) / 256, 256, 0, stream>>>(Wv, Wvt, D, D);
    transpose_cvt_kernel<<<(D * D) / 256, 256, 0, stream>>>(Wo, Wot, D, D);
    transpose_cvt_kernel<<<(D * F) / 256, 256, 0, stream>>>(W1, W1t, D, F);
    transpose_cvt_kernel<<<(F * D) / 256, 256, 0, stream>>>(W2, W2t, F, D);

    // 2) Q/K/V projections (V stored transposed per-head for the PV matmul)
    dim3 blk(256);
    dim3 gQKV((M / BM) * (D / BN));
    gemm_bf16_kernel<<<gQKV, blk, 0, stream>>>(xb, Wqt, bq, Qb,   M, D, D, 0, T, H);
    gemm_bf16_kernel<<<gQKV, blk, 0, stream>>>(xb, Wkt, bk, Kbuf, M, D, D, 0, T, H);
    gemm_bf16_kernel<<<gQKV, blk, 0, stream>>>(xb, Wvt, bv, Vt,   M, D, D, 2, T, H);

    // 3) attention (flash, no mask): 8192 row-tiles / 8 waves per block
    attn_kernel<<<(B * H * (T / 16)) / 8, blk, 0, stream>>>(Qb, Kbuf, Vt, ctxb, T, H, D);

    // 4) output projection -> f32, then residual + LN1
    gemm_bf16_kernel<<<gQKV, blk, 0, stream>>>(ctxb, Wot, bo, raw, M, D, D, 1, T, H);
    ln_kernel<<<M, blk, 0, stream>>>(x, raw, g1, be1, x1f, x1b, D);

    // 5) FFN: relu(x1 @ W1 + b1) @ W2 + b2, residual + LN2 -> d_out
    dim3 gF1((M / BM) * (F / BN));
    gemm_bf16_kernel<<<gF1, blk, 0, stream>>>(x1b, W1t, b1, hb, M, F, D, 3, T, H);
    gemm_bf16_kernel<<<gQKV, blk, 0, stream>>>(hb, W2t, b2, raw, M, D, F, 1, T, H);
    ln_kernel<<<M, blk, 0, stream>>>(x1f, raw, g2, be2, (float*)d_out, (bf16_t*)nullptr, D);
}
